// SimpleMoE_21749714387221
// MI455X (gfx1250) — matrix operand order repaired and checked by reading, not compiled, not run on hardware
//
#include <hip/hip_runtime.h>

typedef __bf16 bf16;
typedef __attribute__((ext_vector_type(16))) __bf16 v16bf;
typedef __attribute__((ext_vector_type(8)))  float  v8f;
typedef __attribute__((ext_vector_type(4)))  int    v4i;

#define WMMA_BF16(A, B, C) \
  __builtin_amdgcn_wmma_f32_16x16x32_bf16(false, (A), false, (B), (short)0, (C), false, false)

constexpr int T_TOK = 4096;   // S*B
constexpr int D_DIM = 1024;
constexpr int F_DIM = 4096;
constexpr int E_NUM = 8;

constexpr int TM = 64;        // tokens per block tile
constexpr int TN = 256;       // output columns per block tile
constexpr int KC = 64;        // K staged per iteration (2 WMMA-K)
constexpr int MAX_TILES = T_TOK / TM;   // 64 (worst case: all tokens on one expert)
constexpr int LDA = 72;       // 144 B rows: 16B-aligned, de-conflicted
constexpr int LDB = 72;

// ---- workspace layout (bytes) ----
constexpr size_t OFF_XB    = 0;                                   // T*D bf16      = 8 MB
constexpr size_t OFF_H     = (size_t)T_TOK * D_DIM * 2;           // 2T*F bf16     = 64 MB
constexpr size_t OFF_TLIST = OFF_H + (size_t)2 * T_TOK * F_DIM * 2;
constexpr size_t OFF_TW    = OFF_TLIST + (size_t)E_NUM * T_TOK * 4;
constexpr size_t OFF_CNT   = OFF_TW + (size_t)E_NUM * T_TOK * 4;
constexpr size_t OFF_OFFS  = OFF_CNT + 64;
constexpr size_t OFF_PSUM  = OFF_OFFS + 64;

// ---------------- CDNA5 async global->LDS copy (ASYNCcnt path) ----------------
#if defined(__has_builtin)
#if __has_builtin(__builtin_amdgcn_global_load_async_to_lds_b128)
#define HAS_ASYNC_LDS 1
#endif
#endif
#ifndef HAS_ASYNC_LDS
#define HAS_ASYNC_LDS 0
#endif

__device__ inline void stage16(void* lds, const void* g) {
#if HAS_ASYNC_LDS
  __builtin_amdgcn_global_load_async_to_lds_b128(
      (__attribute__((address_space(1))) v4i*)(g),
      (__attribute__((address_space(3))) v4i*)(lds), 0, 0);
#else
  *(uint4*)lds = *(const uint4*)g;
#endif
}
__device__ inline void async_fence() {
#if HAS_ASYNC_LDS
  asm volatile("s_wait_asynccnt 0x0" ::: "memory");
#endif
}

// ---------------- fragment loaders (contiguous K = 16*(lane/16) + i, same order for A and B) ----------------
// A: 16x32 bf16 tile, row-major in LDS with leading dim ld.
// lane<16: row=lane,   K = 0..15; lane>=16: row=lane-16, K = 16..31
__device__ inline v16bf load_frag_a(const bf16* sm, int ld) {
  const int lane = threadIdx.x & 31;
  const int row  = lane & 15;
  const int kb   = (lane >> 4) << 4;          // 0 or 16
  const bf16* p  = sm + row * ld + kb;
  v16bf a;
#pragma unroll
  for (int i = 0; i < 16; ++i) a[i] = p[i];
  return a;
}

// B: 32x16 bf16 tile staged TRANSPOSED in LDS as [N][K] with leading dim ld.
// lane<16: col=lane, K=0..15 ; lane>=16: col=lane-16, K=16..31
__device__ inline v16bf load_frag_b(const bf16* smT, int ld) {
  const int lane = threadIdx.x & 31;
  const int n    = lane & 15;
  const int kb   = (lane >> 4) << 4;          // 0 or 16
  const bf16* p  = smT + n * ld + kb;
  v16bf b;
#pragma unroll
  for (int i = 0; i < 16; ++i) b[i] = p[i];
  return b;
}

// ---------------- small kernels ----------------
__global__ __launch_bounds__(256) void moe_cvt_x(const float* __restrict__ x,
                                                 bf16* __restrict__ xb) {
  size_t i = ((size_t)blockIdx.x * 256 + threadIdx.x) * 4;
  float4 v = *(const float4*)(x + i);
  union { bf16 b[4]; uint2 u; } pk;
  pk.b[0] = (bf16)v.x; pk.b[1] = (bf16)v.y; pk.b[2] = (bf16)v.z; pk.b[3] = (bf16)v.w;
  *(uint2*)(xb + i) = pk.u;
}

__global__ __launch_bounds__(256) void moe_gate(const float* __restrict__ x,
                                                const float* __restrict__ Wg,
                                                const float* __restrict__ bg,
                                                int* __restrict__ toklist,
                                                float* __restrict__ tokw,
                                                int* __restrict__ counts,
                                                float* __restrict__ psum) {
  const int wave = threadIdx.x >> 5;
  const int lane = threadIdx.x & 31;
  const int t = blockIdx.x * 8 + wave;

  float part[E_NUM];
#pragma unroll
  for (int e = 0; e < E_NUM; ++e) part[e] = 0.f;
  const float* xr = x + (size_t)t * D_DIM;
  for (int d = lane; d < D_DIM; d += 32) {
    float xv = xr[d];
    const float* wr = Wg + (size_t)d * E_NUM;
#pragma unroll
    for (int e = 0; e < E_NUM; ++e) part[e] += xv * wr[e];
  }
#pragma unroll
  for (int e = 0; e < E_NUM; ++e) {
    float v = part[e];
#pragma unroll
    for (int off = 16; off > 0; off >>= 1) v += __shfl_xor(v, off, 32);
    part[e] = v;
  }

  __shared__ float ps[E_NUM];
  if (threadIdx.x < E_NUM) ps[threadIdx.x] = 0.f;
  __syncthreads();

  if (lane == 0) {
    float p[E_NUM];
    float m = -1e30f;
#pragma unroll
    for (int e = 0; e < E_NUM; ++e) { p[e] = part[e] + bg[e]; m = fmaxf(m, p[e]); }
    float s = 0.f;
#pragma unroll
    for (int e = 0; e < E_NUM; ++e) { p[e] = __expf(p[e] - m); s += p[e]; }
    float inv = 1.f / s;
#pragma unroll
    for (int e = 0; e < E_NUM; ++e) { p[e] *= inv; atomicAdd(&ps[e], p[e]); }
    int e0 = 0;
#pragma unroll
    for (int e = 1; e < E_NUM; ++e) if (p[e] > p[e0]) e0 = e;
    int e1 = (e0 == 0) ? 1 : 0;
#pragma unroll
    for (int e = 0; e < E_NUM; ++e) if (e != e0 && p[e] > p[e1]) e1 = e;
    int pos0 = atomicAdd(&counts[e0], 1);
    toklist[e0 * T_TOK + pos0] = t; tokw[e0 * T_TOK + pos0] = p[e0];
    int pos1 = atomicAdd(&counts[e1], 1);
    toklist[e1 * T_TOK + pos1] = t; tokw[e1 * T_TOK + pos1] = p[e1];
  }
  __syncthreads();
  if (threadIdx.x < E_NUM) atomicAdd(&psum[threadIdx.x], ps[threadIdx.x]);
}

__global__ void moe_prefix_loss(const int* __restrict__ counts,
                                int* __restrict__ offsets,
                                const float* __restrict__ psum,
                                float* __restrict__ loss_out) {
  if (threadIdx.x == 0 && blockIdx.x == 0) {
    int acc = 0;
    for (int e = 0; e < E_NUM; ++e) { offsets[e] = acc; acc += counts[e]; }
    offsets[E_NUM] = acc;
    float L = 0.f;
    for (int e = 0; e < E_NUM; ++e) { float mp = psum[e] / (float)T_TOK; L += mp * mp; }
    loss_out[0] = L * (float)E_NUM;
  }
}

// ---------------- grouped GEMM 1: h = relu(x @ W1 + b1) ----------------
__global__ __launch_bounds__(256) void moe_ffn1(const bf16* __restrict__ xb,
                                                const float* __restrict__ W1,
                                                const float* __restrict__ b1,
                                                const int* __restrict__ toklist,
                                                const int* __restrict__ counts,
                                                const int* __restrict__ offsets,
                                                bf16* __restrict__ h) {
  const int e    = blockIdx.x / MAX_TILES;
  const int tile = blockIdx.x % MAX_TILES;
  const int cnt  = counts[e];
  if (tile * TM >= cnt) return;
  const int fbase = blockIdx.y * TN;

  __shared__ bf16 Asm[TM * LDA];
  __shared__ bf16 Bsm[TN * LDB];
  __shared__ int  tok_s[TM];

  const int tid  = threadIdx.x;
  const int wave = tid >> 5;
  const int mb   = (wave >> 2) * 32;   // wave row block (2x4 wave grid)
  const int nb   = (wave & 3) * 64;    // wave col block

  if (tid < TM) {
    int r = tile * TM + tid;
    if (r >= cnt) r = cnt - 1;                 // pad with a valid token
    tok_s[tid] = toklist[e * T_TOK + r];
  }
  __syncthreads();

  v8f acc[2][4];
  const v8f vzero = {0.f, 0.f, 0.f, 0.f, 0.f, 0.f, 0.f, 0.f};
#pragma unroll
  for (int mi = 0; mi < 2; ++mi)
#pragma unroll
    for (int ni = 0; ni < 4; ++ni) acc[mi][ni] = vzero;

  const float* W1e = W1 + (size_t)e * D_DIM * F_DIM;

  for (int k0 = 0; k0 < D_DIM; k0 += KC) {
    // stage A: 64 tokens x 64 K bf16 (16B chunks, async->LDS when available)
#pragma unroll
    for (int it = 0; it < 2; ++it) {
      int chunk = tid + it * 256;              // 0..511
      int r   = chunk >> 3;
      int col = (chunk & 7) * 8;
      stage16(&Asm[r * LDA + col], xb + (size_t)tok_s[r] * D_DIM + k0 + col);
    }
    // stage B: W1 rows k0..k0+63, cols fbase..fbase+255, fp32->bf16, transposed [f][k]
#pragma unroll
    for (int it = 0; it < 16; ++it) {
      int idx = (tid + it * 256) * 4;          // 0..16380 step 4
      int kk  = idx >> 8;                      // 0..63
      int ff  = idx & 255;
      const float4 v = *(const float4*)(W1e + (size_t)(k0 + kk) * F_DIM + fbase + ff);
      Bsm[(ff + 0) * LDB + kk] = (bf16)v.x;
      Bsm[(ff + 1) * LDB + kk] = (bf16)v.y;
      Bsm[(ff + 2) * LDB + kk] = (bf16)v.z;
      Bsm[(ff + 3) * LDB + kk] = (bf16)v.w;
    }
    async_fence();
    __syncthreads();
#pragma unroll
    for (int kc = 0; kc < KC; kc += 32) {
      v16bf a0 = load_frag_a(&Asm[(mb + 0)  * LDA + kc], LDA);
      v16bf a1 = load_frag_a(&Asm[(mb + 16) * LDA + kc], LDA);
      v16bf bf[4];
#pragma unroll
      for (int ni = 0; ni < 4; ++ni)
        bf[ni] = load_frag_b(&Bsm[(nb + ni * 16) * LDB + kc], LDB);
#pragma unroll
      for (int ni = 0; ni < 4; ++ni) {
        acc[0][ni] = WMMA_BF16(a0, bf[ni], acc[0][ni]);
        acc[1][ni] = WMMA_BF16(a1, bf[ni], acc[1][ni]);
      }
    }
    __syncthreads();
  }

  const int hbase  = offsets[e];
  const int lane   = tid & 31;
  const int rowadd = (lane >> 4) << 3;
  const int ccol   = lane & 15;
  const float* b1e = b1 + (size_t)e * F_DIM;
  float bias[4];
#pragma unroll
  for (int ni = 0; ni < 4; ++ni) bias[ni] = b1e[fbase + nb + ni * 16 + ccol];
#pragma unroll
  for (int mi = 0; mi < 2; ++mi)
#pragma unroll
    for (int v = 0; v < 8; ++v) {
      int m = mb + mi * 16 + rowadd + v;
      int r = tile * TM + m;
      if (r < cnt) {
        bf16* hr = h + (size_t)(hbase + r) * F_DIM + fbase + nb + ccol;
#pragma unroll
        for (int ni = 0; ni < 4; ++ni) {
          float val = fmaxf(acc[mi][ni][v] + bias[ni], 0.f);
          hr[ni * 16] = (bf16)val;
        }
      }
    }
}

// ---------------- grouped GEMM 2: out[tok] += w * (h @ W2 + b2) ----------------
__global__ __launch_bounds__(256) void moe_ffn2(const bf16* __restrict__ h,
                                                const float* __restrict__ W2,
                                                const float* __restrict__ b2,
                                                const int* __restrict__ toklist,
                                                const float* __restrict__ tokw,
                                                const int* __restrict__ counts,
                                                const int* __restrict__ offsets,
                                                float* __restrict__ out) {
  const int e    = blockIdx.x / MAX_TILES;
  const int tile = blockIdx.x % MAX_TILES;
  const int cnt  = counts[e];
  if (tile * TM >= cnt) return;
  const int dbase = blockIdx.y * TN;

  __shared__ bf16 Asm[TM * LDA];
  __shared__ bf16 Bsm[TN * LDB];

  const int tid  = threadIdx.x;
  const int wave = tid >> 5;
  const int mb   = (wave >> 2) * 32;
  const int nb   = (wave & 3) * 64;
  const int hbase = offsets[e];

  v8f acc[2][4];
  const v8f vzero = {0.f, 0.f, 0.f, 0.f, 0.f, 0.f, 0.f, 0.f};
#pragma unroll
  for (int mi = 0; mi < 2; ++mi)
#pragma unroll
    for (int ni = 0; ni < 4; ++ni) acc[mi][ni] = vzero;

  const float* W2e = W2 + (size_t)e * F_DIM * D_DIM;

  for (int k0 = 0; k0 < F_DIM; k0 += KC) {
    // stage A: 64 compact h rows x 64 K (async->LDS when available)
#pragma unroll
    for (int it = 0; it < 2; ++it) {
      int chunk = tid + it * 256;
      int r   = chunk >> 3;
      int col = (chunk & 7) * 8;
      int rr  = tile * TM + r;
      if (rr >= cnt) rr = cnt - 1;
      stage16(&Asm[r * LDA + col], h + (size_t)(hbase + rr) * F_DIM + k0 + col);
    }
    // stage B: W2 rows k0..k0+63, cols dbase..dbase+255, fp32->bf16, transposed
#pragma unroll
    for (int it = 0; it < 16; ++it) {
      int idx = (tid + it * 256) * 4;
      int kk  = idx >> 8;
      int dd  = idx & 255;
      const float4 v = *(const float4*)(W2e + (size_t)(k0 + kk) * D_DIM + dbase + dd);
      Bsm[(dd + 0) * LDB + kk] = (bf16)v.x;
      Bsm[(dd + 1) * LDB + kk] = (bf16)v.y;
      Bsm[(dd + 2) * LDB + kk] = (bf16)v.z;
      Bsm[(dd + 3) * LDB + kk] = (bf16)v.w;
    }
    async_fence();
    __syncthreads();
#pragma unroll
    for (int kc = 0; kc < KC; kc += 32) {
      v16bf a0 = load_frag_a(&Asm[(mb + 0)  * LDA + kc], LDA);
      v16bf a1 = load_frag_a(&Asm[(mb + 16) * LDA + kc], LDA);
      v16bf bf[4];
#pragma unroll
      for (int ni = 0; ni < 4; ++ni)
        bf[ni] = load_frag_b(&Bsm[(nb + ni * 16) * LDB + kc], LDB);
#pragma unroll
      for (int ni = 0; ni < 4; ++ni) {
        acc[0][ni] = WMMA_BF16(a0, bf[ni], acc[0][ni]);
        acc[1][ni] = WMMA_BF16(a1, bf[ni], acc[1][ni]);
      }
    }
    __syncthreads();
  }

  const int lane   = tid & 31;
  const int rowadd = (lane >> 4) << 3;
  const int ccol   = lane & 15;
  const float* b2e = b2 + (size_t)e * D_DIM;
  float bias[4];
#pragma unroll
  for (int ni = 0; ni < 4; ++ni) bias[ni] = b2e[dbase + nb + ni * 16 + ccol];
#pragma unroll
  for (int mi = 0; mi < 2; ++mi)
#pragma unroll
    for (int v = 0; v < 8; ++v) {
      int m = mb + mi * 16 + rowadd + v;
      int r = tile * TM + m;
      if (r < cnt) {
        int tok = toklist[e * T_TOK + r];
        float w = tokw[e * T_TOK + r];
        float* orow = out + (size_t)tok * D_DIM + dbase + nb + ccol;
#pragma unroll
        for (int ni = 0; ni < 4; ++ni) {
          float val = (acc[mi][ni][v] + bias[ni]) * w;
          atomicAdd(&orow[ni * 16], val);
        }
      }
    }
}

extern "C" void kernel_launch(void* const* d_in, const int* in_sizes, int n_in,
                              void* d_out, int out_size, void* d_ws, size_t ws_size,
                              hipStream_t stream) {
  (void)in_sizes; (void)n_in; (void)ws_size;
  const float* x  = (const float*)d_in[0];
  const float* Wg = (const float*)d_in[1];
  const float* bg = (const float*)d_in[2];
  const float* W1 = (const float*)d_in[3];
  const float* b1 = (const float*)d_in[4];
  const float* W2 = (const float*)d_in[5];
  const float* b2 = (const float*)d_in[6];
  float* out = (float*)d_out;

  char* ws = (char*)d_ws;
  bf16*  xb      = (bf16*)(ws + OFF_XB);
  bf16*  h       = (bf16*)(ws + OFF_H);
  int*   toklist = (int*)(ws + OFF_TLIST);
  float* tokw    = (float*)(ws + OFF_TW);
  int*   counts  = (int*)(ws + OFF_CNT);
  int*   offsets = (int*)(ws + OFF_OFFS);
  float* psum    = (float*)(ws + OFF_PSUM);

  (void)hipMemsetAsync(ws + OFF_CNT, 0, 192, stream);
  (void)hipMemsetAsync(d_out, 0, (size_t)out_size * sizeof(float), stream);

  moe_cvt_x<<<(T_TOK * D_DIM) / (256 * 4), 256, 0, stream>>>(x, xb);
  moe_gate<<<T_TOK / 8, 256, 0, stream>>>(x, Wg, bg, toklist, tokw, counts, psum);
  moe_prefix_loss<<<1, 1, 0, stream>>>(counts, offsets, psum, out + (size_t)T_TOK * D_DIM);
  moe_ffn1<<<dim3(E_NUM * MAX_TILES, F_DIM / TN), 256, 0, stream>>>(
      xb, W1, b1, toklist, counts, offsets, h);
  moe_ffn2<<<dim3(E_NUM * MAX_TILES, D_DIM / TN), 256, 0, stream>>>(
      h, W2, b2, toklist, tokw, counts, offsets, out);
}
